// GPT2Attention_88124138979658
// MI455X (gfx1250) — matrix operand order repaired and checked by reading, not compiled, not run on hardware
//
#include <hip/hip_runtime.h>

typedef __attribute__((ext_vector_type(16))) __bf16          v16bf;
typedef __attribute__((ext_vector_type(16))) unsigned short  v16u;
typedef __attribute__((ext_vector_type(8)))  unsigned short  v8u;
typedef __attribute__((ext_vector_type(8)))  float           v8f;
typedef __attribute__((ext_vector_type(4)))  int             v4i;

#define WMMA_BF16(A, B, C) \
  __builtin_amdgcn_wmma_f32_16x16x32_bf16(false, (A), false, (B), (short)0, (C), false, false)

// ---- CDNA5 async global->LDS path (ASYNCcnt-tracked), signature probe-confirmed -----------
#if defined(__has_builtin)
#if __has_builtin(__builtin_amdgcn_global_load_async_to_lds_b128)
#define USE_ASYNC_LDS 1
#endif
#endif

__device__ __forceinline__ void wait_async_lds() {
#ifdef USE_ASYNC_LDS
#if __has_builtin(__builtin_amdgcn_s_wait_asynccnt)
  __builtin_amdgcn_s_wait_asynccnt(0);
#else
  asm volatile("s_wait_asynccnt 0" ::: "memory");
#endif
#endif
}

// copy 16 bf16 (32 bytes) global -> LDS, both contiguous & 32B aligned
__device__ __forceinline__ void copy16_bf16(const unsigned short* __restrict__ g,
                                            unsigned short* l) {
#ifdef USE_ASYNC_LDS
  __builtin_amdgcn_global_load_async_to_lds_b128(
      (__attribute__((address_space(1))) v4i*)g,
      (__attribute__((address_space(3))) v4i*)l, 0, 0);
  __builtin_amdgcn_global_load_async_to_lds_b128(
      (__attribute__((address_space(1))) v4i*)(g + 8),
      (__attribute__((address_space(3))) v4i*)(l + 8), 0, 0);
#else
  const v8u* gp = (const v8u*)g;
  v8u a = gp[0], b = gp[1];
  *(v8u*)l = a;
  *(v8u*)(l + 8) = b;
#endif
}

__device__ __forceinline__ unsigned short f2bf(float f) {
  unsigned int u = __float_as_uint(f);
  u += 0x7FFFu + ((u >> 16) & 1u);          // round-to-nearest-even
  return (unsigned short)(u >> 16);
}

// 16 contiguous bf16 (along N, fixed K row) -> transposed LDS tile (pure copy, no VALU math)
__device__ __forceinline__ void stage16_bf16_T(const unsigned short* __restrict__ g,
                                               unsigned short* l, int stride) {
  const v8u* gp = (const v8u*)g;
  v8u a = gp[0], b = gp[1];
#pragma unroll
  for (int e = 0; e < 8; ++e) {
    l[e * stride]       = a[e];             // ds_store_b16 / _d16_hi
    l[(8 + e) * stride] = b[e];
  }
}

// A fragment: 16x32 bf16 (ISA 7.12.2): lane row = lane&15, K striped over VGPRs/halves.
// Per lane: two contiguous 16B runs -> 2x ds_load_b128.
__device__ __forceinline__ v16bf load_frag_a(const unsigned short* p, int stride, int lane) {
  const int row = lane & 15, hf = lane >> 4;
  const unsigned short* rp = p + row * stride;
  v16u u;
#pragma unroll
  for (int j = 0; j < 8; ++j) {
    const int kidx = (j < 4) ? (hf * 8 + 2 * j) : (16 + hf * 8 + 2 * (j - 4));
    u[2 * j]     = rp[kidx];
    u[2 * j + 1] = rp[kidx + 1];
  }
  return __builtin_bit_cast(v16bf, u);
}

// B fragment 32x16 from a TRANSPOSED tile BT[col][k] (row stride = `stride`):
// per lane two contiguous 16B runs at col*stride + hf*8 and col*stride + 16 + hf*8 -> 2x ds_load_b128.
__device__ __forceinline__ v16bf load_frag_bT(const unsigned short* pT, int stride, int lane) {
  const int col = lane & 15, hf = lane >> 4;
  const unsigned short* rp = pT + col * stride + hf * 8;
  v16u u;
#pragma unroll
  for (int e = 0; e < 16; ++e) u[e] = rp[(e < 8) ? e : (16 + e - 8)];
  return __builtin_bit_cast(v16bf, u);
}

// ---------------- Kernel 0: one-pass fp32 -> bf16 conversion (16 elems/thread) --------------
__global__ __launch_bounds__(128) void cvt_f32_bf16(const float* __restrict__ src,
                                                    unsigned short* __restrict__ dst) {
  const size_t i0 = ((size_t)blockIdx.x * 128 + threadIdx.x) * 16;
  const float4* gp = (const float4*)(src + i0);
  float4 f0 = gp[0], f1 = gp[1], f2 = gp[2], f3 = gp[3];
  v8u a, b;
  a[0] = f2bf(f0.x); a[1] = f2bf(f0.y); a[2] = f2bf(f0.z); a[3] = f2bf(f0.w);
  a[4] = f2bf(f1.x); a[5] = f2bf(f1.y); a[6] = f2bf(f1.z); a[7] = f2bf(f1.w);
  b[0] = f2bf(f2.x); b[1] = f2bf(f2.y); b[2] = f2bf(f2.z); b[3] = f2bf(f2.w);
  b[4] = f2bf(f3.x); b[5] = f2bf(f3.y); b[6] = f2bf(f3.z); b[7] = f2bf(f3.w);
  *(v8u*)(dst + i0) = a;
  *(v8u*)(dst + i0 + 8) = b;
}

// ---------------- Kernel 1: qkv = xb @ Wab + b_attn -> bf16 q/k:[B,H,T,D], v:[B,H,D,T] ------
__global__ __launch_bounds__(128) void qkv_gemm(const unsigned short* __restrict__ xb,
                                                const unsigned short* __restrict__ Wab,
                                                const float* __restrict__ ba,
                                                unsigned short* __restrict__ qg,
                                                unsigned short* __restrict__ kg,
                                                unsigned short* __restrict__ vg) {
  __shared__ unsigned short As[64][32];    // M x K
  __shared__ unsigned short BsT[64][32];   // N x K (transposed)
  const int tid = threadIdx.x, lane = tid & 31, wave = tid >> 5;
  const int mbase = blockIdx.y * 64, nbase = blockIdx.x * 64;
  const int wm = (wave >> 1) * 32, wn = (wave & 1) * 32;
  const int am = tid >> 1, ak0 = (tid & 1) * 16;        // A staging: 16 contiguous K
  const int wk = tid >> 2, wn0 = (tid & 3) * 16;        // B staging: 16 contiguous N
  v8f acc[2][2] = {};

  for (int k0 = 0; k0 < 1024; k0 += 32) {
    __syncthreads();
    copy16_bf16(&xb[(size_t)(mbase + am) * 1024 + k0 + ak0], &As[am][ak0]);
    stage16_bf16_T(&Wab[(size_t)(k0 + wk) * 3072 + nbase + wn0], &BsT[wn0][wk], 32);
    wait_async_lds();
    __syncthreads();
    const v16bf a0 = load_frag_a(&As[wm][0],       32, lane);
    const v16bf a1 = load_frag_a(&As[wm + 16][0],  32, lane);
    const v16bf b0 = load_frag_bT(&BsT[wn][0],      32, lane);
    const v16bf b1 = load_frag_bT(&BsT[wn + 16][0], 32, lane);
    acc[0][0] = WMMA_BF16(a0, b0, acc[0][0]);
    acc[0][1] = WMMA_BF16(a0, b1, acc[0][1]);
    acc[1][0] = WMMA_BF16(a1, b0, acc[1][0]);
    acc[1][1] = WMMA_BF16(a1, b1, acc[1][1]);
  }

  const int col0 = lane & 15, hf = lane >> 4;
#pragma unroll
  for (int i = 0; i < 2; ++i)
#pragma unroll
    for (int j = 0; j < 2; ++j) {
      const int n = nbase + wn + j * 16 + col0;
      const float bias = ba[n];
      const int which = n >> 10;            // 0=q 1=k 2=v (uniform per 16-col tile)
      const int cc = n & 1023, h = cc >> 6, d = cc & 63;
      if (which == 2) {                     // v stored transposed: [B,H,D,T]
#pragma unroll
        for (int r = 0; r < 8; ++r) {
          const int m = mbase + wm + i * 16 + r + 8 * hf;
          const int bb = m >> 11, t = m & 2047;
          vg[((size_t)(bb * 16 + h) * 64 + d) * 2048 + t] = f2bf(acc[i][j][r] + bias);
        }
      } else {
        unsigned short* dst = (which == 0) ? qg : kg;
#pragma unroll
        for (int r = 0; r < 8; ++r) {
          const int m = mbase + wm + i * 16 + r + 8 * hf;
          const int bb = m >> 11, t = m & 2047;
          dst[(((size_t)(bb * 16 + h) * 2048) + t) * 64 + d] = f2bf(acc[i][j][r] + bias);
        }
      }
    }
}

// ---------------- Kernel 2: flash attention, y -> bf16 [B,T,C] ------------------------------
__global__ __launch_bounds__(128) void flash_attn(const unsigned short* __restrict__ qg,
                                                  const unsigned short* __restrict__ kg,
                                                  const unsigned short* __restrict__ vg,
                                                  unsigned short* __restrict__ yg) {
  __shared__ unsigned short Ks[32][64];    // key x d   (B^T for QK^T: contraction over d)
  __shared__ unsigned short VsT[64][32];   // d x key   (B^T for P@V: contraction over key)
  __shared__ unsigned short Ps[4][16][32]; // per-wave P scratch (C-layout -> A-layout)
  const int tid = threadIdx.x, lane = tid & 31, wave = tid >> 5;
  const int bh = blockIdx.x >> 5;
  const int qbase = (blockIdx.x & 31) * 64;
  const int col0 = lane & 15, hf = lane >> 4;
  const size_t base = (size_t)bh * 2048 * 64;
  const int skk = tid >> 2, skd = (tid & 3) * 16;       // K staging coords
  const int svd = tid >> 1, svk = (tid & 1) * 16;       // V staging coords

  // Q fragments (A-layout), loaded once: two contiguous 16B runs per lane per frag
  const int qrow = qbase + wave * 16 + col0;
  v16u qu0, qu1;
#pragma unroll
  for (int j = 0; j < 8; ++j) {
    const int kidx = (j < 4) ? (hf * 8 + 2 * j) : (16 + hf * 8 + 2 * (j - 4));
    qu0[2 * j]     = qg[base + (size_t)qrow * 64 + kidx];
    qu0[2 * j + 1] = qg[base + (size_t)qrow * 64 + kidx + 1];
    qu1[2 * j]     = qg[base + (size_t)qrow * 64 + 32 + kidx];
    qu1[2 * j + 1] = qg[base + (size_t)qrow * 64 + 32 + kidx + 1];
  }
  const v16bf qa0 = __builtin_bit_cast(v16bf, qu0);
  const v16bf qa1 = __builtin_bit_cast(v16bf, qu1);

  v8f o[4] = {};
  float mrow[8], lrow[8];
#pragma unroll
  for (int r = 0; r < 8; ++r) { mrow[r] = -1e30f; lrow[r] = 0.f; }

  const int nkt = (qbase >> 5) + 2;
  for (int kt = 0; kt < nkt; ++kt) {
    const int k0 = kt * 32;
    __syncthreads();                       // protect Ks/VsT from previous iteration readers
    copy16_bf16(&kg[base + (size_t)(k0 + skk) * 64 + skd], &Ks[skk][skd]);
    copy16_bf16(&vg[base + (size_t)svd * 2048 + k0 + svk], &VsT[svd][svk]);
    wait_async_lds();
    __syncthreads();

    // S = Q K^T : two 16x16 tiles; contraction over d in two 32-chunks
    v8f s[2] = {};
#pragma unroll
    for (int n = 0; n < 2; ++n) {
      s[n] = WMMA_BF16(qa0, load_frag_bT(&Ks[n * 16][0],  64, lane), s[n]);
      s[n] = WMMA_BF16(qa1, load_frag_bT(&Ks[n * 16][32], 64, lane), s[n]);
    }

    // scale + causal mask
#pragma unroll
    for (int n = 0; n < 2; ++n) {
      const int key = k0 + n * 16 + col0;
#pragma unroll
      for (int r = 0; r < 8; ++r) {
        const int q = qbase + wave * 16 + r + 8 * hf;
        const float v = s[n][r] * 0.125f;  // 1/sqrt(64)
        s[n][r] = (key > q) ? -1e30f : v;
      }
    }

    // online softmax; row lives across 16 lanes of a half-wave
    float alpha[8];
#pragma unroll
    for (int r = 0; r < 8; ++r) {
      float v = fmaxf(s[0][r], s[1][r]);
#pragma unroll
      for (int off = 8; off >= 1; off >>= 1) v = fmaxf(v, __shfl_xor(v, off, 32));
      const float mn = fmaxf(mrow[r], v);
      alpha[r] = __expf(mrow[r] - mn);
      mrow[r] = mn;
    }
#pragma unroll
    for (int r = 0; r < 8; ++r) {
      const float p0 = __expf(s[0][r] - mrow[r]);
      const float p1 = __expf(s[1][r] - mrow[r]);
      float rs = p0 + p1;
#pragma unroll
      for (int off = 8; off >= 1; off >>= 1) rs += __shfl_xor(rs, off, 32);
      lrow[r] = alpha[r] * lrow[r] + rs;
      const int row = r + 8 * hf;
      Ps[wave][row][col0]      = f2bf(p0);
      Ps[wave][row][16 + col0] = f2bf(p1);
    }
    __syncthreads();

    // O = alpha*O + P @ V
#pragma unroll
    for (int dt = 0; dt < 4; ++dt)
#pragma unroll
      for (int r = 0; r < 8; ++r) o[dt][r] *= alpha[r];

    const v16bf pa = load_frag_a(&Ps[wave][0][0], 32, lane);
#pragma unroll
    for (int dt = 0; dt < 4; ++dt)
      o[dt] = WMMA_BF16(pa, load_frag_bT(&VsT[dt * 16][0], 32, lane), o[dt]);
  }

  // finalize and store y as bf16 in [B,T,C], c = h*64+d
  float inv[8];
#pragma unroll
  for (int r = 0; r < 8; ++r) inv[r] = 1.0f / lrow[r];
  const int bb = bh >> 4, h = bh & 15;
#pragma unroll
  for (int dt = 0; dt < 4; ++dt) {
    const int d = dt * 16 + col0;
#pragma unroll
    for (int r = 0; r < 8; ++r) {
      const int q = qbase + wave * 16 + r + 8 * hf;
      yg[((size_t)(bb * 2048 + q)) * 1024 + h * 64 + d] = f2bf(o[dt][r] * inv[r]);
    }
  }
}

// ---------------- Kernel 3: out = y @ Wpb + b_proj (fp32 out) -------------------------------
__global__ __launch_bounds__(128) void proj_gemm(const unsigned short* __restrict__ yg,
                                                 const unsigned short* __restrict__ Wpb,
                                                 const float* __restrict__ bp,
                                                 float* __restrict__ out) {
  __shared__ unsigned short As[64][32];
  __shared__ unsigned short BsT[64][32];
  const int tid = threadIdx.x, lane = tid & 31, wave = tid >> 5;
  const int mbase = blockIdx.y * 64, nbase = blockIdx.x * 64;
  const int wm = (wave >> 1) * 32, wn = (wave & 1) * 32;
  const int am = tid >> 1, ak0 = (tid & 1) * 16;
  const int wk = tid >> 2, wn0 = (tid & 3) * 16;
  v8f acc[2][2] = {};

  for (int k0 = 0; k0 < 1024; k0 += 32) {
    __syncthreads();
    copy16_bf16(&yg[(size_t)(mbase + am) * 1024 + k0 + ak0], &As[am][ak0]);
    stage16_bf16_T(&Wpb[(size_t)(k0 + wk) * 1024 + nbase + wn0], &BsT[wn0][wk], 32);
    wait_async_lds();
    __syncthreads();
    const v16bf a0 = load_frag_a(&As[wm][0],       32, lane);
    const v16bf a1 = load_frag_a(&As[wm + 16][0],  32, lane);
    const v16bf b0 = load_frag_bT(&BsT[wn][0],      32, lane);
    const v16bf b1 = load_frag_bT(&BsT[wn + 16][0], 32, lane);
    acc[0][0] = WMMA_BF16(a0, b0, acc[0][0]);
    acc[0][1] = WMMA_BF16(a0, b1, acc[0][1]);
    acc[1][0] = WMMA_BF16(a1, b0, acc[1][0]);
    acc[1][1] = WMMA_BF16(a1, b1, acc[1][1]);
  }

  const int col0 = lane & 15, hf = lane >> 4;
#pragma unroll
  for (int i = 0; i < 2; ++i)
#pragma unroll
    for (int j = 0; j < 2; ++j) {
      const int n = nbase + wn + j * 16 + col0;
      const float bias = bp[n];
#pragma unroll
      for (int r = 0; r < 8; ++r) {
        const int m = mbase + wm + i * 16 + r + 8 * hf;
        out[(size_t)m * 1024 + n] = acc[i][j][r] + bias;
      }
    }
}

extern "C" void kernel_launch(void* const* d_in, const int* in_sizes, int n_in,
                              void* d_out, int out_size, void* d_ws, size_t ws_size,
                              hipStream_t stream) {
  (void)in_sizes; (void)n_in; (void)out_size; (void)ws_size;
  const float* x  = (const float*)d_in[0];
  // d_in[1] = attn_mask (bool causal) — implicit in the flash kernel
  const float* Wa = (const float*)d_in[2];
  const float* ba = (const float*)d_in[3];
  const float* Wp = (const float*)d_in[4];
  const float* bp = (const float*)d_in[5];
  float* out = (float*)d_out;

  const size_t HEADS_ELEMS = (size_t)2 * 16 * 2048 * 64;   // B*H*T*D = 4Mi
  unsigned short* qg  = (unsigned short*)d_ws;
  unsigned short* kg  = qg + HEADS_ELEMS;
  unsigned short* vg  = kg + HEADS_ELEMS;                  // [B,H,D,T]
  unsigned short* yg  = vg + HEADS_ELEMS;
  unsigned short* xb  = yg + HEADS_ELEMS;                  // bf16 x     [4096,1024]
  unsigned short* Wab = xb + (size_t)4096 * 1024;          // bf16 Wattn [1024,3072]
  unsigned short* Wpb = Wab + (size_t)1024 * 3072;         // bf16 Wproj [1024,1024]

  // one-pass fp32 -> bf16 (each block converts 128*16 = 2048 elements, exact grids)
  cvt_f32_bf16<<<(4096 * 1024) / 2048, 128, 0, stream>>>(x, xb);
  cvt_f32_bf16<<<(1024 * 3072) / 2048, 128, 0, stream>>>(Wa, Wab);
  cvt_f32_bf16<<<(1024 * 1024) / 2048, 128, 0, stream>>>(Wp, Wpb);

  qkv_gemm<<<dim3(48, 64), 128, 0, stream>>>(xb, Wab, ba, qg, kg, vg);
  flash_attn<<<dim3(2 * 16 * 32), 128, 0, stream>>>(qg, kg, vg, yg);
  proj_gemm<<<dim3(16, 64), 128, 0, stream>>>(yg, Wpb, bp, out);
}
